// CrossAttention_14491219656846
// MI455X (gfx1250) — matrix operand order repaired and checked by reading, not compiled, not run on hardware
//
#include <hip/hip_runtime.h>
#include <hip/hip_bf16.h>

// ---------------------------------------------------------------------------
// CrossAttention for MI455X (gfx1250): f16 WMMA with f32 accumulate.
// B=16, Sq=4096, Skv=77(pad 96), E=1024, H=16, D=64, Dcross=768.
// Weights pre-converted to f16 once; activations converted inline.
// Attention K^T/V tiles staged into LDS via async global->LDS loads.
// ---------------------------------------------------------------------------

typedef _Float16 v16h __attribute__((ext_vector_type(16)));
typedef _Float16 h8v  __attribute__((ext_vector_type(8)));
typedef _Float16 h4v  __attribute__((ext_vector_type(4)));
typedef float    v8f  __attribute__((ext_vector_type(8)));
typedef float    f8v  __attribute__((ext_vector_type(8)));
typedef int      v4i  __attribute__((ext_vector_type(4)));

#define NB   16
#define NH   16
#define NSQ  4096
#define NSKV 77
#define SKVP 96          // padded Skv: 3 full K=32 chunks
#define NE   1024
#define NC   768
#define ND   64

// -------------------- async global->LDS (ASYNCcnt path) --------------------
#if defined(__has_builtin)
#  if __has_builtin(__builtin_amdgcn_global_load_async_to_lds_b128)
#    define HAVE_ASYNC_LDS 1
#  endif
#endif
#ifndef HAVE_ASYNC_LDS
#  define HAVE_ASYNC_LDS 0
#endif

typedef __attribute__((address_space(1))) v4i* gv4i_p;   // global 128-bit
typedef __attribute__((address_space(3))) v4i* lv4i_p;   // LDS 128-bit

__device__ __forceinline__ void ca_async_copy16(const void* g, void* l) {
#if HAVE_ASYNC_LDS
    __builtin_amdgcn_global_load_async_to_lds_b128(
        (gv4i_p)(g), (lv4i_p)(l), 0, 0);
#else
    *(float4*)l = *(const float4*)g;   // fallback: through VGPRs
#endif
}
__device__ __forceinline__ void ca_async_wait0() {
#if HAVE_ASYNC_LDS
#  if __has_builtin(__builtin_amdgcn_s_wait_asynccnt)
    __builtin_amdgcn_s_wait_asynccnt(0);
#  else
    asm volatile("s_wait_asynccnt 0" ::: "memory");
#  endif
#endif
}

__device__ __forceinline__ v8f wmma16(v16h a, v16h b, v8f c) {
    // D = A(16x32 f16) x B(32x16 f16) + C(16x16 f32)
    return __builtin_amdgcn_wmma_f32_16x16x32_f16(
        /*neg_a=*/false, a, /*neg_b=*/false, b,
        /*c_mod=*/(short)0, c, /*reuse_a=*/false, /*reuse_b=*/false);
}

// A-operand (16x32, 16-bit) per-lane load. ISA layout:
// lanes 0-15: row M=lane, K = 0..7 (VGPR0-3) and 16..23 (VGPR4-7)
// lanes16-31: row M=lane-16, K = 8..15 and 24..31
// -> per lane: 8 contiguous halves at kbase, 8 more at kbase+16.
__device__ __forceinline__ v16h load_a_f32(const float* p0) {
    f8v lo = *(const f8v*)p0;
    f8v hi = *(const f8v*)(p0 + 16);
    v16h a;
#pragma unroll
    for (int i = 0; i < 8; ++i) { a[i] = (_Float16)lo[i]; a[i + 8] = (_Float16)hi[i]; }
    return a;
}
__device__ __forceinline__ v16h load_a_f16(const _Float16* p0) {
    h8v lo = *(const h8v*)p0;
    h8v hi = *(const h8v*)(p0 + 16);
    v16h a;
#pragma unroll
    for (int i = 0; i < 8; ++i) { a[i] = lo[i]; a[i + 8] = hi[i]; }
    return a;
}

// ---------------------------------------------------------------------------
// fp32 -> f16 conversion (weights), 4 elements / thread. n % 4 == 0.
// ---------------------------------------------------------------------------
__global__ void ca_cvt_f16_kernel(const float* __restrict__ in,
                                  _Float16* __restrict__ out, int n) {
    int i = (blockIdx.x * blockDim.x + threadIdx.x) * 4;
    if (i < n) {
        float4 v = *(const float4*)(in + i);
        h4v h;
        h[0] = (_Float16)v.x; h[1] = (_Float16)v.y;
        h[2] = (_Float16)v.z; h[3] = (_Float16)v.w;
        *(h4v*)(out + i) = h;
    }
}

// ---------------------------------------------------------------------------
// Zero the padded K^T / V staging buffers (6.29 MB contiguous).
// ---------------------------------------------------------------------------
__global__ void ca_zero_kernel(float4* __restrict__ p, int n) {
    int i = blockIdx.x * blockDim.x + threadIdx.x;
    if (i < n) p[i] = make_float4(0.f, 0.f, 0.f, 0.f);
}

// ---------------------------------------------------------------------------
// Q projection: Q = f16(x @ wq + bq), stored [B*Sq, E] f16.
// One wave: 16 rows x 128 cols (8 C tiles). waves = 4096*8 = 32768.
// The 8 waves of a block share the same 16 x-rows (A reuse via WGP$).
// ---------------------------------------------------------------------------
__global__ __launch_bounds__(256) void ca_proj_q_kernel(
    const float* __restrict__ X, const _Float16* __restrict__ Wh,
    const float* __restrict__ bias, _Float16* __restrict__ Q) {
    const int tid  = threadIdx.x;
    const int lane = tid & 31;
    const int wave = blockIdx.x * 8 + (tid >> 5);
    const int ncg  = wave & 7;            // column group of 128
    const int rt   = wave >> 3;           // row tile of 16
    const int col0 = ncg * 128;
    const int m    = lane & 15;
    const int kb   = (lane >> 4) * 8;
    const size_t rowbase = (size_t)(rt * 16 + m) * NE;

    v8f c[8] = {{}, {}, {}, {}, {}, {}, {}, {}};
    for (int k0 = 0; k0 < NE; k0 += 32) {
        const float* ap = X + rowbase + k0 + kb;
        __builtin_prefetch(ap + 32, 0, 1);             // global_prefetch_b8
        v16h a = load_a_f32(ap);
        const _Float16* bp = Wh + (size_t)(k0 + kb) * NE + col0 + m;
#pragma unroll
        for (int ct = 0; ct < 8; ++ct) {
            v16h bm;
            for (int i = 0; i < 8; ++i) { bm[i] = bp[(size_t)i * NE + ct * 16]; bm[i + 8] = bp[(size_t)(i + 16) * NE + ct * 16]; }
            c[ct] = wmma16(a, bm, c[ct]);
        }
    }
    const int n   = lane & 15;
    const int rhi = (lane >> 4) * 8;
#pragma unroll
    for (int ct = 0; ct < 8; ++ct) {
        const int col = col0 + ct * 16 + n;
        const float bv = bias[col];
#pragma unroll
        for (int v = 0; v < 8; ++v) {
            const size_t r = (size_t)(rt * 16 + v + rhi);
            Q[r * NE + col] = (_Float16)(c[ct][v] + bv);
        }
    }
}

// ---------------------------------------------------------------------------
// K/V projection from y[B,77,768] with f16 weights. isK=1: store K^T
// [B,H,D,SKVP]; isK=0: store V [B,H,SKVP,D]. Rows >= 77 not stored
// (buffer pre-zeroed). waves = B(16)*mtiles(5)*ncg(16) = 1280 -> 160 blocks.
// ---------------------------------------------------------------------------
__global__ __launch_bounds__(256) void ca_proj_kv_kernel(
    const float* __restrict__ Y, const _Float16* __restrict__ Wh,
    const float* __restrict__ bias, _Float16* __restrict__ Out, int isK) {
    const int tid  = threadIdx.x;
    const int lane = tid & 31;
    const int wave = blockIdx.x * 8 + (tid >> 5);
    const int ncg  = wave % 16;
    const int mt   = (wave / 16) % 5;
    const int b    = wave / 80;
    const int col0 = ncg * 64;
    const int m    = lane & 15;
    const int kb   = (lane >> 4) * 8;
    const int row  = mt * 16 + m;
    const int rowc = row < NSKV ? row : (NSKV - 1);   // clamp loads, skip stores
    const size_t rowbase = ((size_t)b * NSKV + rowc) * NC;

    v8f c[4] = {{}, {}, {}, {}};
    for (int k0 = 0; k0 < NC; k0 += 32) {
        v16h a = load_a_f32(Y + rowbase + k0 + kb);
        const _Float16* bp = Wh + (size_t)(k0 + kb) * NE + col0 + m;
#pragma unroll
        for (int ct = 0; ct < 4; ++ct) {
            v16h bm;
            for (int i = 0; i < 8; ++i) { bm[i] = bp[(size_t)i * NE + ct * 16]; bm[i + 8] = bp[(size_t)(i + 16) * NE + ct * 16]; }
            c[ct] = wmma16(a, bm, c[ct]);
        }
    }
    const int n   = lane & 15;
    const int rhi = (lane >> 4) * 8;
#pragma unroll
    for (int ct = 0; ct < 4; ++ct) {
        const int col = col0 + ct * 16 + n;
        const float bv = bias[col];
        const int h = col >> 6;       // head
        const int d = col & 63;       // dim within head
#pragma unroll
        for (int v = 0; v < 8; ++v) {
            const int kv = mt * 16 + v + rhi;
            if (kv < NSKV) {
                _Float16 val = (_Float16)(c[ct][v] + bv);
                if (isK)  // K^T: [b,h,d,kv]
                    Out[(((size_t)b * NH + h) * ND + d) * SKVP + kv] = val;
                else      // V: [b,h,kv,d]
                    Out[(((size_t)b * NH + h) * SKVP + kv) * ND + d] = val;
            }
        }
    }
}

// ---------------------------------------------------------------------------
// Attention: per wave 16 q-rows x full Skv x D. waves = 16*16*256 = 65536.
// All 8 waves of a block share (b,h): K^T (12KB) and V (12KB) tiles are
// staged into LDS once per block with ASYNC global->LDS loads, then used
// as WMMA B-operands straight from LDS. scores via WMMA (Q x K^T), f32
// softmax with half-wave shuffles, probs staged through LDS
// (C-layout -> A-layout), then P x V via WMMA.
// ---------------------------------------------------------------------------
__global__ __launch_bounds__(256) void ca_attn_kernel(
    const _Float16* __restrict__ Q, const _Float16* __restrict__ Kt,
    const _Float16* __restrict__ V, _Float16* __restrict__ AO) {
    __shared__ __align__(32) _Float16 KtLds[ND][SKVP];    // 12 KB
    __shared__ __align__(32) _Float16 VLds[SKVP][ND];     // 12 KB
    __shared__ __align__(32) _Float16 Plds[8][16][SKVP];  // 24 KB

    const int tid  = threadIdx.x;
    const int lane = tid & 31;
    const int widx = tid >> 5;
    const int wave = blockIdx.x * 8 + widx;
    const int qt   = wave & 255;          // q tile of 16 (Sq/16 = 256)
    const int h    = (wave >> 8) & 15;
    const int b    = wave >> 12;

    const int m  = lane & 15;
    const int kb = (lane >> 4) * 8;
    const size_t grow = ((size_t)b * NSQ + qt * 16 + m) * NE + h * ND;
    const size_t bh   = (size_t)b * NH + h;

    // ---- Cooperative async stage of K^T and V tiles (12288 B each) ----
    {
        const char* ktg = (const char*)(Kt + bh * (ND * SKVP));
        const char* vg  = (const char*)(V  + bh * (SKVP * ND));
        char* ktl = (char*)&KtLds[0][0];
        char* vl  = (char*)&VLds[0][0];
#pragma unroll
        for (int i = 0; i < 3; ++i) {              // 768 x 16B chunks / 256 thr
            const int idx = (tid + i * 256) * 16;
            ca_async_copy16(ktg + idx, ktl + idx);
            ca_async_copy16(vg + idx, vl + idx);
        }
        ca_async_wait0();
    }
    __syncthreads();

    // A-operands for scores: Q rows, d-chunks 0..31 and 32..63.
    v16h a0 = load_a_f16(Q + grow + 0  + kb);
    v16h a1 = load_a_f16(Q + grow + 32 + kb);

    // Scores: 5 kv tiles of 16 (kv 0..79), K-dim = D = 2 chunks of 32.
    v8f s[5] = {{}, {}, {}, {}, {}};
#pragma unroll
    for (int t = 0; t < 5; ++t) {
        v16h kb0, kb1;
        for (int i = 0; i < 8; ++i) { kb0[i] = KtLds[kb + i][t * 16 + m]; kb0[i + 8] = KtLds[kb + 16 + i][t * 16 + m]; kb1[i] = KtLds[32 + kb + i][t * 16 + m]; kb1[i + 8] = KtLds[48 + kb + i][t * 16 + m]; }
        s[t] = wmma16(a0, kb0, s[t]);
        s[t] = wmma16(a1, kb1, s[t]);
    }

    // Scale, mask padded kv columns (77..79 live in tile 4).
    const int n = lane & 15;
#pragma unroll
    for (int t = 0; t < 5; ++t)
#pragma unroll
        for (int i = 0; i < 8; ++i) s[t][i] *= 0.125f;   // 1/sqrt(64)
    if (64 + n >= NSKV) {
#pragma unroll
        for (int i = 0; i < 8; ++i) s[4][i] = -1.0e30f;
    }

    // Row softmax. C-layout: VGPR v -> row (v | v+8), N = lane%16; row
    // reductions = xor-shuffles with masks < 16 (stay inside each half).
    const int rhi = (lane >> 4) * 8;
#pragma unroll
    for (int v = 0; v < 8; ++v) {
        float rmax = -3.0e38f;
#pragma unroll
        for (int t = 0; t < 5; ++t) rmax = fmaxf(rmax, s[t][v]);
        for (int off = 1; off < 16; off <<= 1)
            rmax = fmaxf(rmax, __shfl_xor(rmax, off, 32));
        float pv[5], rsum = 0.f;
#pragma unroll
        for (int t = 0; t < 5; ++t) { pv[t] = __expf(s[t][v] - rmax); rsum += pv[t]; }
        for (int off = 1; off < 16; off <<= 1)
            rsum += __shfl_xor(rsum, off, 32);
        const float inv = 1.0f / rsum;
        const int row = v + rhi;
#pragma unroll
        for (int t = 0; t < 5; ++t)
            Plds[widx][row][t * 16 + n] = (_Float16)(pv[t] * inv);
    }
    // Zero pad probs kv 80..95 (V padding rows are zero anyway).
    {
        const int c0 = 80 + rhi;
#pragma unroll
        for (int i = 0; i < 8; ++i) Plds[widx][m][c0 + i] = (_Float16)0.f;
    }
    __syncthreads();

    // Out = P(16xSKVP) x V(SKVPxD): 3 K-chunks x 4 d-tiles, V from LDS.
    v8f o[4] = {{}, {}, {}, {}};
#pragma unroll
    for (int kc = 0; kc < 3; ++kc) {
        v16h pa = load_a_f16(&Plds[widx][m][kc * 32 + kb]);
#pragma unroll
        for (int dt = 0; dt < 4; ++dt) {
            v16h vb;
            for (int i = 0; i < 8; ++i) { vb[i] = VLds[kc * 32 + kb + i][dt * 16 + m]; vb[i + 8] = VLds[kc * 32 + kb + 16 + i][dt * 16 + m]; }
            o[dt] = wmma16(pa, vb, o[dt]);
        }
    }
#pragma unroll
    for (int dt = 0; dt < 4; ++dt) {
#pragma unroll
        for (int v = 0; v < 8; ++v) {
            const size_t r = (size_t)b * NSQ + qt * 16 + v + rhi;
            AO[r * NE + h * ND + dt * 16 + n] = (_Float16)o[dt][v];
        }
    }
}

// ---------------------------------------------------------------------------
// Output projection: out = f32(attn @ wo + bo), A and W both f16.
// One wave: 16 rows x 128 cols. waves = 4096*8 = 32768 -> 4096 blocks.
// ---------------------------------------------------------------------------
__global__ __launch_bounds__(256) void ca_proj_o_kernel(
    const _Float16* __restrict__ A, const _Float16* __restrict__ Wh,
    const float* __restrict__ bias, float* __restrict__ Out) {
    const int tid  = threadIdx.x;
    const int lane = tid & 31;
    const int wave = blockIdx.x * 8 + (tid >> 5);
    const int ncg  = wave & 7;
    const int rt   = wave >> 3;
    const int col0 = ncg * 128;
    const int m    = lane & 15;
    const int kb   = (lane >> 4) * 8;
    const size_t rowbase = (size_t)(rt * 16 + m) * NE;

    v8f c[8] = {{}, {}, {}, {}, {}, {}, {}, {}};
    for (int k0 = 0; k0 < NE; k0 += 32) {
        const _Float16* ap = A + rowbase + k0 + kb;
        __builtin_prefetch(ap + 32, 0, 1);
        v16h a = load_a_f16(ap);
        const _Float16* bp = Wh + (size_t)(k0 + kb) * NE + col0 + m;
#pragma unroll
        for (int ct = 0; ct < 8; ++ct) {
            v16h bm;
            for (int i = 0; i < 8; ++i) { bm[i] = bp[(size_t)i * NE + ct * 16]; bm[i + 8] = bp[(size_t)(i + 16) * NE + ct * 16]; }
            c[ct] = wmma16(a, bm, c[ct]);
        }
    }
    const int n   = lane & 15;
    const int rhi = (lane >> 4) * 8;
#pragma unroll
    for (int ct = 0; ct < 8; ++ct) {
        const int col = col0 + ct * 16 + n;
        const float bv = bias[col];
#pragma unroll
        for (int v = 0; v < 8; ++v) {
            const size_t r = (size_t)(rt * 16 + v + rhi);
            Out[r * NE + col] = c[ct][v] + bv;
        }
    }
}

// ---------------------------------------------------------------------------
extern "C" void kernel_launch(void* const* d_in, const int* in_sizes, int n_in,
                              void* d_out, int out_size, void* d_ws, size_t ws_size,
                              hipStream_t stream) {
    const float* x  = (const float*)d_in[0];
    const float* y  = (const float*)d_in[1];
    const float* wq = (const float*)d_in[2];
    const float* bq = (const float*)d_in[3];
    const float* wk = (const float*)d_in[4];
    const float* bk = (const float*)d_in[5];
    const float* wv = (const float*)d_in[6];
    const float* bv = (const float*)d_in[7];
    const float* wo = (const float*)d_in[8];
    const float* bo = (const float*)d_in[9];
    float* out = (float*)d_out;

    // Workspace layout (f16):
    // Q [B*Sq,E] | Kt [B,H,D,SKVP] | V [B,H,SKVP,D] | AO [B*Sq,E]
    // | wqh [E,E] | wkh [C,E] | wvh [C,E] | woh [E,E]
    char* ws = (char*)d_ws;
    const size_t qBytes  = (size_t)NB * NSQ * NE * 2;            // 134 MB
    const size_t ktBytes = (size_t)NB * NH * ND * SKVP * 2;      // 3 MB
    const size_t wEE     = (size_t)NE * NE;                      // elements
    const size_t wCE     = (size_t)NC * NE;                      // elements
    size_t off = 0;
    _Float16* Q   = (_Float16*)(ws + off); off += qBytes;
    _Float16* Kt  = (_Float16*)(ws + off); off += ktBytes;
    _Float16* Vb  = (_Float16*)(ws + off); off += ktBytes;
    _Float16* AO  = (_Float16*)(ws + off); off += qBytes;
    _Float16* wqh = (_Float16*)(ws + off); off += wEE * 2;
    _Float16* wkh = (_Float16*)(ws + off); off += wCE * 2;
    _Float16* wvh = (_Float16*)(ws + off); off += wCE * 2;
    _Float16* woh = (_Float16*)(ws + off); off += wEE * 2;

    // 0. One-time weight conversion fp32 -> f16 (L2-resident, ~7 MB).
    {
        int nq = (int)wEE, nc = (int)wCE;
        ca_cvt_f16_kernel<<<(nq / 4 + 255) / 256, 256, 0, stream>>>(wq, wqh, nq);
        ca_cvt_f16_kernel<<<(nc / 4 + 255) / 256, 256, 0, stream>>>(wk, wkh, nc);
        ca_cvt_f16_kernel<<<(nc / 4 + 255) / 256, 256, 0, stream>>>(wv, wvh, nc);
        ca_cvt_f16_kernel<<<(nq / 4 + 255) / 256, 256, 0, stream>>>(wo, woh, nq);
    }
    // 1. Zero padded K^T / V staging (contiguous 2*ktBytes).
    {
        int nvec = (int)((2 * ktBytes) / 16);
        ca_zero_kernel<<<(nvec + 255) / 256, 256, 0, stream>>>((float4*)Kt, nvec);
    }
    // 2. K and V projections (1280 waves -> 160 blocks each).
    ca_proj_kv_kernel<<<160, 256, 0, stream>>>(y, wkh, bk, Kt, 1);
    ca_proj_kv_kernel<<<160, 256, 0, stream>>>(y, wvh, bv, Vb, 0);
    // 3. Q projection (32768 waves -> 4096 blocks).
    ca_proj_q_kernel<<<4096, 256, 0, stream>>>(x, wqh, bq, Q);
    // 4. Attention (65536 waves -> 8192 blocks).
    ca_attn_kernel<<<8192, 256, 0, stream>>>(Q, Kt, Vb, AO);
    // 5. Output projection (32768 waves -> 4096 blocks).
    ca_proj_o_kernel<<<4096, 256, 0, stream>>>(AO, woh, bo, out);
}
